// SGFormerAdapted_14877766714051
// MI455X (gfx1250) — matrix operand order repaired and checked by reading, not compiled, not run on hardware
//
#include <hip/hip_runtime.h>
#include <hip/hip_bf16.h>

typedef __attribute__((ext_vector_type(16))) __bf16       v16bf;
typedef __attribute__((ext_vector_type(8)))  float        v8f;
typedef __attribute__((ext_vector_type(4)))  float        v4f;
typedef __attribute__((ext_vector_type(8)))  unsigned int v8u;

#define N_NODES 100000
#define N_EDGES 800000
#define D_HID   256
#define D_OUT   64
#define LDSA_STRIDE 34
#define LDSB_STRIDE 34

__device__ __forceinline__ unsigned int pack_bf16(float a, float b) {
    union { unsigned int u; __bf16 h[2]; } c;
    c.h[0] = (__bf16)a;
    c.h[1] = (__bf16)b;
    return c.u;
}

// ---------------------------------------------------------------------------
// Fragment loaders per CDNA5 ISA 7.12.2 (16-bit A 16x32) and 7.12.4 (B layout)
// ---------------------------------------------------------------------------
__device__ __forceinline__ v16bf load_a_frag(const __bf16* lds, int row_base, int lane) {
    const int half = lane >> 4;
    const int m    = lane & 15;
    v8u u;
#pragma unroll
    for (int v = 0; v < 8; ++v) {
        int kb = ((v & 3) * 2) + ((v >> 2) * 16) + half * 8;   // K pair index
        u[v] = *(const unsigned int*)&lds[(row_base + m) * LDSA_STRIDE + kb];
    }
    return __builtin_bit_cast(v16bf, u);
}

__device__ __forceinline__ v16bf load_b_frag(const __bf16* lds, int col_base, int lane) {
    const int half = lane >> 4;
    const int n    = lane & 15;
    v8u u;
#pragma unroll
    for (int v = 0; v < 8; ++v) {
        int kb = ((v & 3) * 2) + ((v >> 2) * 16) + half * 8;   // K pair index
        u[v] = *(const unsigned int*)&lds[(col_base + n) * LDSB_STRIDE + kb];
    }
    return __builtin_bit_cast(v16bf, u);
}

// ---------------------------------------------------------------------------
// C[M x Nc] = A[M x K] @ W[K x Nc] (+bias). fp32 in HBM -> bf16 in LDS ->
// v_wmma_f32_16x16x32_bf16, fp32 accumulate. Software-pipelined: global loads
// for tile k+1 are in flight (registers) while tile k's WMMAs execute.
// Block 256 thr = 8 waves; tile 128(M) x 64(Nc); K multiple of 32.
// ---------------------------------------------------------------------------
__global__ __launch_bounds__(256)
void gemm_wmma(const float* __restrict__ A, const float* __restrict__ W,
               const float* __restrict__ bias, float* __restrict__ C,
               int M, int K, int Nc, const float* __restrict__ scaleA_ss) {
    __shared__ __bf16 lds_a[128 * LDSA_STRIDE];
    __shared__ __bf16 lds_b[64 * LDSB_STRIDE];
    const int t = threadIdx.x, lane = t & 31, wave = t >> 5;
    const int row0 = blockIdx.x * 128;
    const int col0 = blockIdx.y * 64;
    const float sA = scaleA_ss ? rsqrtf(*scaleA_ss) : 1.0f;

    // per-thread staging coordinates (A: 1024 float4 slots; B: 512 float4 slots)
    const int a_m[4]  = {(t + 0) >> 3, (t + 256) >> 3, (t + 512) >> 3, (t + 768) >> 3};
    const int a_kk4   = (t & 7) * 4;
    const int b_kk[2] = {(t + 0) >> 4, (t + 256) >> 4};
    const int b_d4    = (t & 15) * 4;

    // loop-invariant clamped row + mask (branchless tail handling)
    size_t a_row_off[4];
    float  a_msk[4];
#pragma unroll
    for (int i = 0; i < 4; ++i) {
        int gr = row0 + a_m[i];
        int grc = (gr < M) ? gr : (M - 1);
        a_row_off[i] = (size_t)grc * K;
        a_msk[i] = (gr < M) ? sA : 0.0f;
    }

    v8f zero8 = {0.f, 0.f, 0.f, 0.f, 0.f, 0.f, 0.f, 0.f};
    v8f acc[4] = {zero8, zero8, zero8, zero8};

    v4f areg[4], breg[2];
#pragma unroll
    for (int i = 0; i < 4; ++i) areg[i] = *(const v4f*)&A[a_row_off[i] + a_kk4];
#pragma unroll
    for (int i = 0; i < 2; ++i) breg[i] = *(const v4f*)&W[(size_t)b_kk[i] * Nc + col0 + b_d4];

    for (int k0 = 0; k0 < K; k0 += 32) {
        __syncthreads();
        // commit current tile registers to LDS (bf16)
#pragma unroll
        for (int i = 0; i < 4; ++i) {
            unsigned int* dst = (unsigned int*)&lds_a[a_m[i] * LDSA_STRIDE + a_kk4];
            dst[0] = pack_bf16(areg[i][0] * a_msk[i], areg[i][1] * a_msk[i]);
            dst[1] = pack_bf16(areg[i][2] * a_msk[i], areg[i][3] * a_msk[i]);
        }
#pragma unroll
        for (int i = 0; i < 2; ++i)
#pragma unroll
            for (int j = 0; j < 4; ++j)
                lds_b[(b_d4 + j) * LDSB_STRIDE + b_kk[i]] = (__bf16)breg[i][j];
        __syncthreads();

        // issue next tile's global loads; they complete during the WMMAs below
        int k0n = (k0 + 32 < K) ? (k0 + 32) : k0;
#pragma unroll
        for (int i = 0; i < 4; ++i) areg[i] = *(const v4f*)&A[a_row_off[i] + k0n + a_kk4];
#pragma unroll
        for (int i = 0; i < 2; ++i)
            breg[i] = *(const v4f*)&W[(size_t)(k0n + b_kk[i]) * Nc + col0 + b_d4];

        v16bf afrag = load_a_frag(lds_a, wave * 16, lane);
        v16bf bfrag[4];
#pragma unroll
        for (int dt = 0; dt < 4; ++dt) bfrag[dt] = load_b_frag(lds_b, dt * 16, lane);
#pragma unroll
        for (int dt = 0; dt < 4; ++dt)
            acc[dt] = __builtin_amdgcn_wmma_f32_16x16x32_bf16(
                false, afrag, false, bfrag[dt], (short)0, acc[dt], false, false);
    }
    // C/D layout: VGPR r -> M = r + 8*(lane>=16); N = lane&15
    const int msub = (lane >> 4) * 8;
    const int coll = lane & 15;
#pragma unroll
    for (int dt = 0; dt < 4; ++dt) {
#pragma unroll
        for (int r = 0; r < 8; ++r) {
            int gm = row0 + wave * 16 + msub + r;
            if (gm < M) {
                int gc = col0 + dt * 16 + coll;
                float val = acc[dt][r];
                if (bias) val += bias[gc];
                C[(size_t)gm * Nc + gc] = val;
            }
        }
    }
}

// ---------------------------------------------------------------------------
// kv[256x256] += sum_n (K[n,m]*rsqrt(*kss)) * V[n,d]  -- split-K over grid.z,
// fp32 atomic reduction into the 256x256 output (stays in L2). Same register
// double-buffered pipeline; masks ride along with the pipelined registers.
// ---------------------------------------------------------------------------
__global__ __launch_bounds__(256)
void gemm_ktv_wmma(const float* __restrict__ Kmat, const float* __restrict__ Vmat,
                   float* __restrict__ Cout, int Nrows, const float* __restrict__ scaleA_ss) {
    __shared__ __bf16 lds_a[128 * LDSA_STRIDE];
    __shared__ __bf16 lds_b[64 * LDSB_STRIDE];
    const int t = threadIdx.x, lane = t & 31, wave = t >> 5;
    const int m0 = blockIdx.x * 128, d0 = blockIdx.y * 64;
    const int nsplit = gridDim.z;
    const int chunk = (Nrows + nsplit - 1) / nsplit;
    const int nstart = blockIdx.z * chunk;
    const int nend = (nstart + chunk < Nrows) ? (nstart + chunk) : Nrows;
    const float sA = scaleA_ss ? rsqrtf(*scaleA_ss) : 1.0f;

    // A: lds_a[m][kk] = K[n+kk][m0+m]; 1024 float4 slots, m-fast (coalesced)
    const int a_kk[4] = {(t + 0) >> 5, (t + 256) >> 5, (t + 512) >> 5, (t + 768) >> 5};
    const int a_m4    = (t & 31) * 4;
    const int b_kk[2] = {(t + 0) >> 4, (t + 256) >> 4};
    const int b_d4    = (t & 15) * 4;

    v8f zero8 = {0.f, 0.f, 0.f, 0.f, 0.f, 0.f, 0.f, 0.f};
    v8f acc[4] = {zero8, zero8, zero8, zero8};

    v4f areg[4], breg[2];
    float amsk[4], bmsk[2];
    auto load_tile = [&](int n) {
#pragma unroll
        for (int i = 0; i < 4; ++i) {
            int gn = n + a_kk[i];
            int gnc = (gn < nend) ? gn : (nend - 1);
            amsk[i] = (gn < nend) ? sA : 0.0f;
            areg[i] = *(const v4f*)&Kmat[(size_t)gnc * 256 + m0 + a_m4];
        }
#pragma unroll
        for (int i = 0; i < 2; ++i) {
            int gn = n + b_kk[i];
            int gnc = (gn < nend) ? gn : (nend - 1);
            bmsk[i] = (gn < nend) ? 1.0f : 0.0f;
            breg[i] = *(const v4f*)&Vmat[(size_t)gnc * 256 + d0 + b_d4];
        }
    };
    load_tile(nstart);

    for (int n = nstart; n < nend; n += 32) {
        __syncthreads();
#pragma unroll
        for (int i = 0; i < 4; ++i)
#pragma unroll
            for (int j = 0; j < 4; ++j)
                lds_a[(a_m4 + j) * LDSA_STRIDE + a_kk[i]] = (__bf16)(areg[i][j] * amsk[i]);
#pragma unroll
        for (int i = 0; i < 2; ++i)
#pragma unroll
            for (int j = 0; j < 4; ++j)
                lds_b[(b_d4 + j) * LDSB_STRIDE + b_kk[i]] = (__bf16)(breg[i][j] * bmsk[i]);
        __syncthreads();

        int nn = (n + 32 < nend) ? (n + 32) : n;
        load_tile(nn);                             // overlaps with WMMAs below

        v16bf afrag = load_a_frag(lds_a, wave * 16, lane);
        v16bf bfrag[4];
#pragma unroll
        for (int dt = 0; dt < 4; ++dt) bfrag[dt] = load_b_frag(lds_b, dt * 16, lane);
#pragma unroll
        for (int dt = 0; dt < 4; ++dt)
            acc[dt] = __builtin_amdgcn_wmma_f32_16x16x32_bf16(
                false, afrag, false, bfrag[dt], (short)0, acc[dt], false, false);
    }
    const int msub = (lane >> 4) * 8;
    const int coll = lane & 15;
#pragma unroll
    for (int dt = 0; dt < 4; ++dt)
#pragma unroll
        for (int r = 0; r < 8; ++r)
            atomicAdd(&Cout[(size_t)(m0 + wave * 16 + msub + r) * 256 + d0 + dt * 16 + coll],
                      acc[dt][r]);
}

// ---------------------------------------------------------------------------
// Fused per-row epilogues (one block per row of 256)
// ---------------------------------------------------------------------------
__device__ __forceinline__ float block_reduce_sum(float x, float* red, int c) {
    red[c] = x;
    __syncthreads();
    for (int s = 128; s > 0; s >>= 1) {
        if (c < s) red[c] += red[c + s];
        __syncthreads();
    }
    float r = red[0];
    __syncthreads();
    return r;
}

__global__ __launch_bounds__(256)
void ln_act_kernel(const float* __restrict__ in, const float* __restrict__ g,
                   const float* __restrict__ b, float* __restrict__ out, int relu) {
    __shared__ float red[256];
    const int row = blockIdx.x, c = threadIdx.x;
    const size_t o = (size_t)row * 256 + c;
    float x = in[o];
    float mu = block_reduce_sum(x, red, c) * (1.0f / 256.0f);
    float d = x - mu;
    float var = block_reduce_sum(d * d, red, c) * (1.0f / 256.0f);
    float y = d * rsqrtf(var + 1e-5f) * g[c] + b[c];
    if (relu) y = fmaxf(y, 0.0f);
    out[o] = y;
}

// h <- LN(0.5 * ((qkv + N*v)/normalizer) + 0.5 * h)
__global__ __launch_bounds__(256)
void attn_epilogue(const float* __restrict__ qkv, const float* __restrict__ v,
                   const float* __restrict__ q, const float* __restrict__ qss,
                   const float* __restrict__ ksum, const float* __restrict__ g,
                   const float* __restrict__ b, float* __restrict__ h) {
    __shared__ float red[256];
    const int row = blockIdx.x, c = threadIdx.x;
    const size_t o = (size_t)row * 256 + c;
    const float invq = rsqrtf(*qss);
    float normalizer = block_reduce_sum(q[o] * invq * ksum[c], red, c) + (float)N_NODES;
    float num = qkv[o] + (float)N_NODES * v[o];
    float z = 0.5f * (num / normalizer) + 0.5f * h[o];
    float mu = block_reduce_sum(z, red, c) * (1.0f / 256.0f);
    float d = z - mu;
    float var = block_reduce_sum(d * d, red, c) * (1.0f / 256.0f);
    h[o] = d * rsqrtf(var + 1e-5f) * g[c] + b[c];
}

__global__ void sumsq_kernel(const float* __restrict__ x, size_t n, float* __restrict__ out) {
    __shared__ float red[256];
    float s = 0.0f;
    for (size_t i = (size_t)blockIdx.x * blockDim.x + threadIdx.x; i < n;
         i += (size_t)gridDim.x * blockDim.x) {
        float v = x[i];
        s += v * v;
    }
    float tot = block_reduce_sum(s, red, threadIdx.x);
    if (threadIdx.x == 0) atomicAdd(out, tot);
}

__global__ __launch_bounds__(256)
void ksum_kernel(const float* __restrict__ k, const float* __restrict__ kss,
                 float* __restrict__ ksum) {
    __shared__ float red[256];
    const int c = blockIdx.x;
    float s = 0.0f;
    for (int n = threadIdx.x; n < N_NODES; n += 256) s += k[(size_t)n * 256 + c];
    float tot = block_reduce_sum(s, red, threadIdx.x);
    if (threadIdx.x == 0) ksum[c] = tot * rsqrtf(*kss);
}

// ---------------------------------------------------------------------------
// Utility / GCN kernels
// ---------------------------------------------------------------------------
__global__ void fill_kernel(float* __restrict__ p, size_t n, float v) {
    size_t i = (size_t)blockIdx.x * blockDim.x + threadIdx.x;
    if (i < n) p[i] = v;
}

__global__ void deg_edges_kernel(const long long* __restrict__ dst, float* __restrict__ deg) {
    size_t e = (size_t)blockIdx.x * blockDim.x + threadIdx.x;
    if (e < N_EDGES) atomicAdd(&deg[(int)dst[e]], 1.0f);
}

__global__ void dinv_kernel(const float* __restrict__ deg, float* __restrict__ dinv) {
    size_t i = (size_t)blockIdx.x * blockDim.x + threadIdx.x;
    if (i < N_NODES) dinv[i] = rsqrtf(deg[i]);   // deg >= 1 (self loop)
}

// g[n,c] = dinv[n]^2 * h[n,c]   (self-loop term)
__global__ void self_loop_kernel(const float* __restrict__ h, const float* __restrict__ dinv,
                                 float* __restrict__ g) {
    size_t i = (size_t)blockIdx.x * blockDim.x + threadIdx.x;
    if (i < (size_t)N_NODES * 256) {
        int n = (int)(i >> 8);
        float w = dinv[n];
        g[i] = w * w * h[i];
    }
}

// one thread per (edge, 4 columns): g[dst] += dinv[s]*dinv[d] * h[src]
__global__ void edge_scatter_kernel(const long long* __restrict__ src,
                                    const long long* __restrict__ dst,
                                    const float* __restrict__ dinv,
                                    const float* __restrict__ h, float* __restrict__ g) {
    size_t tid = (size_t)blockIdx.x * blockDim.x + threadIdx.x;
    if (tid >= (size_t)N_EDGES * 64) return;
    int e = (int)(tid >> 6);
    int c4 = (int)(tid & 63) * 4;
    int s = (int)src[e], d = (int)dst[e];
    float w = dinv[s] * dinv[d];
    const float* hs = h + (size_t)s * 256 + c4;
    float* gd = g + (size_t)d * 256 + c4;
    atomicAdd(&gd[0], w * hs[0]);
    atomicAdd(&gd[1], w * hs[1]);
    atomicAdd(&gd[2], w * hs[2]);
    atomicAdd(&gd[3], w * hs[3]);
}

// g <- relu((g + b0) * (bn_g/sqrt(1+eps)) + bn_b)
__global__ void bn_relu_kernel(float* __restrict__ g, const float* __restrict__ b0,
                               const float* __restrict__ bng, const float* __restrict__ bnb) {
    size_t i = (size_t)blockIdx.x * blockDim.x + threadIdx.x;
    if (i < (size_t)N_NODES * 256) {
        int c = (int)(i & 255);
        float s = bng[c] * rsqrtf(1.0f + 1e-5f);
        g[i] = fmaxf((g[i] + b0[c]) * s + bnb[c], 0.0f);
    }
}

// emb = 0.8*(gagg + b1) + 0.2*h_trans
__global__ void combine_kernel(const float* __restrict__ gagg, const float* __restrict__ b1,
                               const float* __restrict__ h, float* __restrict__ emb) {
    size_t i = (size_t)blockIdx.x * blockDim.x + threadIdx.x;
    if (i < (size_t)N_NODES * 256) {
        int c = (int)(i & 255);
        emb[i] = 0.8f * (gagg[i] + b1[c]) + 0.2f * h[i];
    }
}

// ---------------------------------------------------------------------------
// Host orchestration
// ---------------------------------------------------------------------------
extern "C" void kernel_launch(void* const* d_in, const int* in_sizes, int n_in,
                              void* d_out, int out_size, void* d_ws, size_t ws_size,
                              hipStream_t stream) {
    (void)in_sizes; (void)n_in; (void)out_size; (void)ws_size;
    const float* x      = (const float*)d_in[0];
    const long long* ei = (const long long*)d_in[1];   // int64 [2, E]
    const long long* src = ei;
    const long long* dst = ei + N_EDGES;
    const float* fc0_w = (const float*)d_in[2];  const float* fc0_b = (const float*)d_in[3];
    const float* ln0_g = (const float*)d_in[4];  const float* ln0_b = (const float*)d_in[5];
    const float* lw[2][8];
    for (int l = 0; l < 2; ++l)
        for (int j = 0; j < 8; ++j) lw[l][j] = (const float*)d_in[6 + l * 8 + j];
    // lw[l] = {qw, qb, kw, kb, vw, vb, ln_g, ln_b}
    const float* gcn_w0 = (const float*)d_in[22]; const float* gcn_b0 = (const float*)d_in[23];
    const float* bn_g   = (const float*)d_in[24]; const float* bn_b   = (const float*)d_in[25];
    const float* gcn_w1 = (const float*)d_in[26]; const float* gcn_b1 = (const float*)d_in[27];
    const float* fc_w   = (const float*)d_in[28]; const float* fc_b   = (const float*)d_in[29];
    float* out = (float*)d_out;

    const size_t NB = (size_t)N_NODES * 256;           // elements per activation buffer
    char* ws = (char*)d_ws;
    size_t off = 0;
    auto alloc = [&](size_t elems) { float* p = (float*)(ws + off); off += ((elems * 4 + 255) & ~(size_t)255); return p; };
    float* h    = alloc(NB);
    float* tmp  = alloc(NB);
    float* q    = alloc(NB);
    float* k    = alloc(NB);
    float* v    = alloc(NB);
    float* deg  = alloc(N_NODES);
    float* dinv = alloc(N_NODES);
    float* kv   = alloc(256 * 256);
    float* ksum = alloc(256);
    float* scal = alloc(8);                            // scal[0]=||q||^2, scal[1]=||k||^2

    const dim3 blk(256);
    const int GM = (N_NODES + 127) / 128;              // 782
    const dim3 g256(GM, 4), g64(GM, 1), gktv(2, 4, 32);
    const int EW_BLOCKS = (int)((NB + 255) / 256);     // elementwise grids
    const int E_BLOCKS  = (N_EDGES + 255) / 256;
    const int ES_BLOCKS = (int)(((size_t)N_EDGES * 64 + 255) / 256);

    // ---- TransConv stem: h = relu(LN(x @ fc0 + b)) ----
    gemm_wmma<<<g256, blk, 0, stream>>>(x, fc0_w, fc0_b, tmp, N_NODES, 256, 256, nullptr);
    ln_act_kernel<<<N_NODES, blk, 0, stream>>>(tmp, ln0_g, ln0_b, h, 1);

    // ---- Two attention layers ----
    for (int l = 0; l < 2; ++l) {
        gemm_wmma<<<g256, blk, 0, stream>>>(h, lw[l][0], lw[l][1], q, N_NODES, 256, 256, nullptr);
        gemm_wmma<<<g256, blk, 0, stream>>>(h, lw[l][2], lw[l][3], k, N_NODES, 256, 256, nullptr);
        gemm_wmma<<<g256, blk, 0, stream>>>(h, lw[l][4], lw[l][5], v, N_NODES, 256, 256, nullptr);
        fill_kernel<<<1, 8, 0, stream>>>(scal, 8, 0.0f);
        sumsq_kernel<<<4096, blk, 0, stream>>>(q, NB, &scal[0]);
        sumsq_kernel<<<4096, blk, 0, stream>>>(k, NB, &scal[1]);
        ksum_kernel<<<256, blk, 0, stream>>>(k, &scal[1], ksum);
        fill_kernel<<<256, blk, 0, stream>>>(kv, 256 * 256, 0.0f);
        gemm_ktv_wmma<<<gktv, blk, 0, stream>>>(k, v, kv, N_NODES, &scal[1]);
        gemm_wmma<<<g256, blk, 0, stream>>>(q, kv, nullptr, tmp, N_NODES, 256, 256, &scal[0]);
        attn_epilogue<<<N_NODES, blk, 0, stream>>>(tmp, v, q, &scal[0], ksum,
                                                   lw[l][6], lw[l][7], h);
    }

    // ---- GCN branch ----
    fill_kernel<<<(N_NODES + 255) / 256, blk, 0, stream>>>(deg, N_NODES, 1.0f);
    deg_edges_kernel<<<E_BLOCKS, blk, 0, stream>>>(dst, deg);
    dinv_kernel<<<(N_NODES + 255) / 256, blk, 0, stream>>>(deg, dinv);

    gemm_wmma<<<g256, blk, 0, stream>>>(x, gcn_w0, nullptr, tmp, N_NODES, 256, 256, nullptr);
    self_loop_kernel<<<EW_BLOCKS, blk, 0, stream>>>(tmp, dinv, q);
    edge_scatter_kernel<<<ES_BLOCKS, blk, 0, stream>>>(src, dst, dinv, tmp, q);
    bn_relu_kernel<<<EW_BLOCKS, blk, 0, stream>>>(q, gcn_b0, bn_g, bn_b);

    gemm_wmma<<<g256, blk, 0, stream>>>(q, gcn_w1, nullptr, tmp, N_NODES, 256, 256, nullptr);
    self_loop_kernel<<<EW_BLOCKS, blk, 0, stream>>>(tmp, dinv, k);
    edge_scatter_kernel<<<ES_BLOCKS, blk, 0, stream>>>(src, dst, dinv, tmp, k);

    // ---- combine + classifier ----
    combine_kernel<<<EW_BLOCKS, blk, 0, stream>>>(k, gcn_b1, h, tmp);
    gemm_wmma<<<g64, blk, 0, stream>>>(tmp, fc_w, fc_b, out, N_NODES, 256, D_OUT, nullptr);
}
